// LSTMTaggerPlus_19945828122824
// MI455X (gfx1250) — matrix operand order repaired and checked by reading, not compiled, not run on hardware
//
#include <hip/hip_runtime.h>
#include <hip/hip_bf16.h>
#include <math.h>

// ---------------------------------------------------------------------------
// LSTMTagger on MI455X (gfx1250, wave32, WMMA) — latency-oriented schedule.
//
//  K0: char_w_hh f32 -> f16 (grid-wide, off critical path)
//  K1: block 0   : Xg[16,1024] = char_e @ w_ih^T via WMMA, then 16 sequential
//                  LSTM steps with the recurrent matvec h @ w_hh^T on the
//                  matrix pipe (f16 operands, f32 accumulate, h/c f32 in LDS).
//      blocks 1-8: gates2 = word_w_ih[:, :512] @ word_e + biases via WMMA —
//                  runs CONCURRENTLY with the recurrence (no dependency).
//  K2: gates2 += word_w_ih[:, 512:768] @ char_h (only 2MB serialized).
//  K3: word nonlinearity (c0=0 => c=i*g), tag logits, log-softmax.
//
// Matvec-as-WMMA trick: only C row M=0 is ever read, and WMMA rows are
// independent, so A rows 1..15 may hold garbage. All lanes therefore load the
// SAME vector values (LDS broadcast) — no masking, no divergent loads.
// A fragments are fetched as two ds_load_b128 from f16-staged LDS (the
// contiguous k order used here (k = 16h + s) makes each lane-half's 16 K values adjacent).
// ---------------------------------------------------------------------------

typedef _Float16     v16h __attribute__((ext_vector_type(16)));
typedef _Float16     v8h  __attribute__((ext_vector_type(8)));
typedef _Float16     h4_t __attribute__((ext_vector_type(4)));
typedef float        v8f  __attribute__((ext_vector_type(8)));
typedef unsigned int v8u  __attribute__((ext_vector_type(8)));

#define CLEN   16
#define CEMB   128
#define CHID   256
#define CG     (4*CHID)        // 1024 char gate rows
#define WEMB   512
#define WHID   512
#define WG4    (4*WHID)        // 2048 word gate rows
#define WIN    (WEMB+CHID)     // 768 word-LSTM input
#define NTAG   50

__device__ __forceinline__ float sigm(float x) { return 1.0f / (1.0f + __expf(-x)); }

// Build one f16 A-fragment from packed-f16 LDS (uint view), K-chunk kk.
// Layout (contiguous): VGPR v -> K = kk*32 + sub*16 + 2v.
// In uint units: base = kk*16 + sub*8, elements {base..base+7}.
__device__ __forceinline__ v16h load_a_frag(const unsigned int* u, int kk, int sub) {
  const int base = kk * 16 + sub * 8;
  uint4 q0 = *(const uint4*)(u + base);
  uint4 q1 = *(const uint4*)(u + base + 4);
  v8u a = { q0.x, q0.y, q0.z, q0.w, q1.x, q1.y, q1.z, q1.w };
  return __builtin_bit_cast(v16h, a);
}

// ---------------------------------------------------------------------------
// K0: convert char_w_hh [1024,256] f32 -> f16 (row-major preserved).
// ---------------------------------------------------------------------------
__global__ __launch_bounds__(256) void k_cvt_whh(const float* __restrict__ w,
                                                 _Float16* __restrict__ out) {
  const int i = (blockIdx.x * 256 + threadIdx.x) * 4;   // 256 blocks cover 1024*256
  float4 v = *(const float4*)(w + i);
  h4_t o = { (_Float16)v.x, (_Float16)v.y, (_Float16)v.z, (_Float16)v.w };
  *(h4_t*)(out + i) = o;
}

// ---------------------------------------------------------------------------
// K1: 9 blocks x 512 threads.
// ---------------------------------------------------------------------------
__global__ __launch_bounds__(512) void k_main(
    const int* __restrict__ word_seq, const int* __restrict__ char_seq,
    const float* __restrict__ word_emb, const float* __restrict__ char_emb,
    const float* __restrict__ c_w_ih,   // [1024,128]
    const _Float16* __restrict__ wh16,  // [1024,256] (from K0)
    const float* __restrict__ c_b_ih, const float* __restrict__ c_b_hh,
    const float* __restrict__ w_w_ih,   // [2048,768]
    const float* __restrict__ w_b_ih, const float* __restrict__ w_b_hh,
    float* __restrict__ Xg,             // ws: [16][1024]
    float* __restrict__ charh,          // ws: [256]
    float* __restrict__ gates2)         // ws: [2048]
{
  __shared__ _Float16 ceh[CLEN][CEMB];   // 4KB  char embeddings (f16)
  __shared__ float    h_s[CHID], c_s[CHID];
  __shared__ _Float16 h16s[CHID];        // f16 staging of h for A-fragments
  __shared__ float    gates[CG];
  __shared__ _Float16 xw16[WEMB];        // word path: word_e staged as f16

  const int tid  = threadIdx.x;
  const int wave = tid >> 5;
  const int lane = tid & 31;
  const int sub  = lane >> 4;
  const int lr   = lane & 15;

  if (blockIdx.x == 0) {
    // ================= char pipeline (critical path) =================
    for (int i = tid; i < CLEN * CEMB; i += 512) {
      int t = i >> 7, k = i & (CEMB - 1);
      ceh[t][k] = (_Float16)char_emb[char_seq[t] * CEMB + k];
    }
    for (int i = tid; i < CHID; i += 512) {
      h_s[i] = 0.0f; c_s[i] = 0.0f; h16s[i] = (_Float16)0.0f;
    }
    __syncthreads();

    // ---- Phase A: Xg = char_e @ c_w_ih^T + b  (real 16-row A, 64 N-tiles) ----
    for (int nt = wave; nt < CG / 16; nt += 16) {
      v8f acc = {};
#pragma unroll
      for (int kk = 0; kk < CEMB / 32; ++kk) {
        v16h A = load_a_frag((const unsigned int*)ceh[lr], kk, sub);  // row M=lr
        v16h B;
        const float2* wr = (const float2*)(c_w_ih + (nt * 16 + lr) * CEMB + kk * 32 + sub * 16);
#pragma unroll
        for (int v = 0; v < 8; ++v) {
          float2 p = wr[v];
          B[2 * v]     = (_Float16)p.x;
          B[2 * v + 1] = (_Float16)p.y;
        }
        acc = __builtin_amdgcn_wmma_f32_16x16x32_f16(false, A, false, B,
                                                     (short)0, acc, false, false);
      }
#pragma unroll
      for (int r = 0; r < 8; ++r) {
        const int m   = r + sub * 8;
        const int col = nt * 16 + lr;
        Xg[m * CG + col] = acc[r] + c_b_ih[col] + c_b_hh[col];
      }
    }
    __threadfence();
    __syncthreads();

    // ---- Phase B: 16 sequential steps; recurrence on the matrix pipe ----
    for (int t = 0; t < CLEN; ++t) {
      // A fragments: every lane loads the same h values (broadcast); rows
      // 1..15 of A are harmless replicas (only C row 0 is read).
      v16h Afr[8];
#pragma unroll
      for (int kk = 0; kk < CHID / 32; ++kk)
        Afr[kk] = load_a_frag((const unsigned int*)h16s, kk, sub);

#pragma unroll 2
      for (int ntl = 0; ntl < 4; ++ntl) {        // 16 waves x 4 = 64 N-tiles
        const int nt = wave * 4 + ntl;
        v8f acc = {};
#pragma unroll
        for (int kk = 0; kk < CHID / 32; ++kk) {
          const v8h* bp = (const v8h*)(wh16 + (nt * 16 + lr) * CHID + kk * 32 + sub * 16);
          v8h b0 = bp[0], b1 = bp[1];
          v16h B = __builtin_shufflevector(b0, b1, 0, 1, 2, 3, 4, 5, 6, 7,
                                           8, 9, 10, 11, 12, 13, 14, 15);
          acc = __builtin_amdgcn_wmma_f32_16x16x32_f16(false, Afr[kk], false, B,
                                                       (short)0, acc, false, false);
        }
        if (sub == 0) {                          // M=0 -> VGPR0, lanes 0-15
          const int col = nt * 16 + lr;
          gates[col] = acc[0] + Xg[t * CG + col];
        }
      }
      __syncthreads();
      if (tid < CHID) {                          // torch gate order i,f,g,o
        const float ig = sigm(gates[tid]);
        const float fg = sigm(gates[CHID + tid]);
        const float gg = tanhf(gates[2 * CHID + tid]);
        const float og = sigm(gates[3 * CHID + tid]);
        const float c  = fg * c_s[tid] + ig * gg;
        const float hn = og * tanhf(c);
        c_s[tid]  = c;
        h_s[tid]  = hn;
        h16s[tid] = (_Float16)hn;                // f16 staging for next step's A
      }
      __syncthreads();
    }
    if (tid < CHID) charh[tid] = h_s[tid];
  } else {
    // ========== word_e partial (overlaps recurrence; no dependency) ==========
    for (int i = tid; i < WEMB; i += 512)
      xw16[i] = (_Float16)word_emb[word_seq[0] * WEMB + i];
    __syncthreads();

    const int nt = (blockIdx.x - 1) * 16 + wave;   // 128 N-tiles over 8 blocks
    v8f acc = {};
#pragma unroll 4
    for (int kk = 0; kk < WEMB / 32; ++kk) {       // K = 512 over word_e slice
      v16h A = load_a_frag((const unsigned int*)xw16, kk, sub);  // broadcast
      const float* wrow = w_w_ih + (nt * 16 + lr) * WIN;         // cols 0..511
      const float2* wr  = (const float2*)(wrow + kk * 32 + sub * 16);
      __builtin_prefetch(wrow + kk * 32 + 128, 0, 1);
      v16h B;
#pragma unroll
      for (int v = 0; v < 8; ++v) {
        float2 p = wr[v];
        B[2 * v]     = (_Float16)p.x;
        B[2 * v + 1] = (_Float16)p.y;
      }
      acc = __builtin_amdgcn_wmma_f32_16x16x32_f16(false, A, false, B,
                                                   (short)0, acc, false, false);
    }
    if (sub == 0) {
      const int col = nt * 16 + lr;
      gates2[col] = acc[0] + w_b_ih[col] + w_b_hh[col];
    }
  }
}

// ---------------------------------------------------------------------------
// K2: gates2 += word_w_ih[:, 512:768] @ char_h  (2MB, the only serialized part)
// ---------------------------------------------------------------------------
__global__ __launch_bounds__(512) void k_word_charpart(
    const float* __restrict__ w_ih,      // [2048,768]
    const float* __restrict__ charh,     // [256]
    float* __restrict__ gates2)
{
  __shared__ _Float16 x16[CHID];
  const int tid = threadIdx.x;
  for (int i = tid; i < CHID; i += 512) x16[i] = (_Float16)charh[i];
  __syncthreads();

  const int wave = tid >> 5, lane = tid & 31;
  const int sub  = lane >> 4, lr = lane & 15;
  const int nt   = blockIdx.x * 16 + wave;

  v8f acc = {};
#pragma unroll
  for (int kk = 0; kk < CHID / 32; ++kk) {
    v16h A = load_a_frag((const unsigned int*)x16, kk, sub);    // broadcast
    const float* wrow = w_ih + (nt * 16 + lr) * WIN + WEMB;     // cols 512..767
    const float2* wr  = (const float2*)(wrow + kk * 32 + sub * 16);
    __builtin_prefetch(wrow + kk * 32 + 64, 0, 1);
    v16h B;
#pragma unroll
    for (int v = 0; v < 8; ++v) {
      float2 p = wr[v];
      B[2 * v]     = (_Float16)p.x;
      B[2 * v + 1] = (_Float16)p.y;
    }
    acc = __builtin_amdgcn_wmma_f32_16x16x32_f16(false, A, false, B,
                                                 (short)0, acc, false, false);
  }
  if (sub == 0) {
    const int col = nt * 16 + lr;
    gates2[col] = gates2[col] + acc[0];
  }
}

// ---------------------------------------------------------------------------
// K3: word nonlinearity (c0=0 => c=i*g), tag logits, log-softmax.
// ---------------------------------------------------------------------------
__global__ __launch_bounds__(512) void k_final(
    const float* __restrict__ gates2,
    const float* __restrict__ tag_w,  // [50,512]
    const float* __restrict__ tag_b,
    float* __restrict__ out)          // [50]
{
  __shared__ float h[WHID];
  __shared__ float logits[64];
  __shared__ float red[2];
  const int tid = threadIdx.x;

  {
    const float ig = sigm(gates2[tid]);
    const float gg = tanhf(gates2[2 * WHID + tid]);
    const float og = sigm(gates2[3 * WHID + tid]);
    h[tid] = og * tanhf(ig * gg);              // f*c0 == 0
  }
  __syncthreads();

  const int row = tid >> 3, part = tid & 7;    // 8 lanes per tag row
  float s = 0.0f;
  if (row < NTAG) {
    const float4* wr = (const float4*)(tag_w + row * WHID + part * 64);
    const float*  hp = h + part * 64;
#pragma unroll 4
    for (int k = 0; k < 16; ++k) {
      float4 w4 = wr[k];
      s += w4.x * hp[4 * k]     + w4.y * hp[4 * k + 1]
         + w4.z * hp[4 * k + 2] + w4.w * hp[4 * k + 3];
    }
  }
#pragma unroll
  for (int off = 1; off < 8; off <<= 1) s += __shfl_xor(s, off, 32);
  if (part == 0 && row < NTAG) logits[row] = s + tag_b[row];
  __syncthreads();

  if (tid == 0) {
    float mx = -INFINITY;
    for (int i = 0; i < NTAG; ++i) mx = fmaxf(mx, logits[i]);
    float sum = 0.0f;
    for (int i = 0; i < NTAG; ++i) sum += __expf(logits[i] - mx);
    red[0] = mx;
    red[1] = logf(sum);
  }
  __syncthreads();
  if (tid < NTAG) out[tid] = logits[tid] - red[0] - red[1];
}

// ---------------------------------------------------------------------------
extern "C" void kernel_launch(void* const* d_in, const int* in_sizes, int n_in,
                              void* d_out, int out_size, void* d_ws, size_t ws_size,
                              hipStream_t stream) {
  const int*   word_seq  = (const int*)  d_in[0];
  const int*   char_seq  = (const int*)  d_in[1];
  const float* word_emb  = (const float*)d_in[2];
  const float* char_emb  = (const float*)d_in[3];
  const float* c_w_ih    = (const float*)d_in[4];
  const float* c_w_hh    = (const float*)d_in[5];
  const float* c_b_ih    = (const float*)d_in[6];
  const float* c_b_hh    = (const float*)d_in[7];
  const float* w_w_ih    = (const float*)d_in[8];
  // d_in[9] word_w_hh unused: word LSTM starts from zero state.
  const float* w_b_ih    = (const float*)d_in[10];
  const float* w_b_hh    = (const float*)d_in[11];
  const float* tag_w     = (const float*)d_in[12];
  const float* tag_b     = (const float*)d_in[13];

  char* ws = (char*)d_ws;
  _Float16* wh16   = (_Float16*)ws;                     // 1024*256 f16 = 512KB
  float*    Xg     = (float*)(ws + 512 * 1024);         // 16*1024 f32  = 64KB
  float*    charh  = Xg + CLEN * CG;                    // 256 f32
  float*    gates2 = charh + CHID;                      // 2048 f32

  k_cvt_whh<<<256, 256, 0, stream>>>(c_w_hh, wh16);
  k_main<<<9, 512, 0, stream>>>(word_seq, char_seq, word_emb, char_emb,
                                c_w_ih, wh16, c_b_ih, c_b_hh,
                                w_w_ih, w_b_ih, w_b_hh,
                                Xg, charh, gates2);
  k_word_charpart<<<WG4 / 256, 512, 0, stream>>>(w_w_ih, charh, gates2);
  k_final<<<1, 512, 0, stream>>>(gates2, tag_w, tag_b, (float*)d_out);
}
